// CWCT_54022098649493
// MI455X (gfx1250) — compile-verified, matrix operand order checked by reading, not run on hardware
//
#include <hip/hip_runtime.h>

typedef __attribute__((ext_vector_type(16))) __bf16 v16bf;
typedef __attribute__((ext_vector_type(8)))  float  v8f;

#define NCH 256
#define HW  (512*512)
#define NL  8
#define MP  (HW + 64*NL)   // padded compacted width (each label segment 64-aligned)
#define KSPLIT 8
#define ASTR 40            // padded LDS row stride (bf16 elems) to stagger banks

__device__ __forceinline__ unsigned short f2bf(float f){
  unsigned u = __float_as_uint(f);
  u += 0x7FFFu + ((u >> 16) & 1u);          // round-to-nearest-even
  return (unsigned short)(u >> 16);
}
__device__ __forceinline__ float bf2f(unsigned short h){
  return __uint_as_float(((unsigned)h) << 16);
}

// ---------------- histogram of labels ----------------
__global__ void k_hist(const int* __restrict__ seg, int* __restrict__ cnt){
  __shared__ int h[NL];
  if (threadIdx.x < NL) h[threadIdx.x] = 0;
  __syncthreads();
  for (int i = blockIdx.x*blockDim.x + threadIdx.x; i < HW; i += gridDim.x*blockDim.x)
    atomicAdd(&h[seg[i] & (NL-1)], 1);
  __syncthreads();
  if (threadIdx.x < NL) atomicAdd(&cnt[threadIdx.x], h[threadIdx.x]);
}

// ---------------- padded offsets + validity ----------------
__global__ void k_offsets(const int* __restrict__ cnt_c, const int* __restrict__ cnt_s,
                          int* __restrict__ off_c, int* __restrict__ off_s,
                          int* __restrict__ padc_c, int* __restrict__ padc_s,
                          int* __restrict__ valid, int* __restrict__ totals){
  if (threadIdx.x != 0 || blockIdx.x != 0) return;
  int oc = 0, os = 0;
  for (int l = 0; l < NL; ++l){
    int pc = (cnt_c[l] + 63) & ~63;
    int ps = (cnt_s[l] + 63) & ~63;
    off_c[l] = oc; off_s[l] = os; padc_c[l] = pc; padc_s[l] = ps;
    oc += pc; os += ps;
    float nc = (float)cnt_c[l], ns = (float)cnt_s[l];
    valid[l] = (nc > 10.f) && (ns > 10.f) && (nc < 100.f*ns) && (ns < 100.f*nc);
  }
  totals[0] = oc; totals[1] = os;
}

// ---------------- compute compacted destination per pixel ----------------
__global__ void k_scatter(const int* __restrict__ seg, int* __restrict__ cur,
                          const int* __restrict__ off, int* __restrict__ dst,
                          int* __restrict__ idx){
  int i = blockIdx.x*blockDim.x + threadIdx.x;
  if (i >= HW) return;
  int l = seg[i] & (NL-1);
  int pos = atomicAdd(&cur[l], 1);
  int j = off[l] + pos;
  dst[i] = j;
  if (idx) idx[j] = i;
}

// ---------------- zero the pad columns (<=63 per label) ----------------
__global__ void k_zeropad(unsigned short* __restrict__ Xc, unsigned short* __restrict__ Xp,
                          unsigned short* __restrict__ Xs,
                          int* __restrict__ idx_c,
                          const int* __restrict__ cnt_c, const int* __restrict__ off_c,
                          const int* __restrict__ padc_c,
                          const int* __restrict__ cnt_s, const int* __restrict__ off_s,
                          const int* __restrict__ padc_s){
  int l = blockIdx.x, tx = threadIdx.x;
  for (int c = cnt_c[l] + tx; c < padc_c[l]; c += blockDim.x){
    int col = off_c[l] + c;
    idx_c[col] = -1;
    for (int ch = 0; ch < NCH; ++ch){
      Xc[(size_t)ch*MP + col] = 0;
      Xp[(size_t)col*NCH + ch] = 0;
    }
  }
  for (int c = cnt_s[l] + tx; c < padc_s[l]; c += blockDim.x){
    int col = off_s[l] + c;
    for (int ch = 0; ch < NCH; ++ch) Xs[(size_t)ch*MP + col] = 0;
  }
}

// ---------------- gather features into compacted bf16 matrices ----------------
// Xc: channel-major (for Gram/colsum), Xp: pixel-major (for apply, optional)
__global__ void k_gather(const float* __restrict__ feat, const int* __restrict__ dst,
                         unsigned short* __restrict__ X, unsigned short* __restrict__ Xp){
  int p  = blockIdx.x*blockDim.x + threadIdx.x;
  int ch = blockIdx.y;
  if (p >= HW) return;
  unsigned short v = f2bf(feat[(size_t)ch*HW + p]);
  int j = dst[p];
  X[(size_t)ch*MP + j] = v;
  if (Xp) Xp[(size_t)j*NCH + ch] = v;
}

// ---------------- per-label per-channel sums (means) ----------------
__global__ void k_colsum(const unsigned short* __restrict__ X,
                         const int* __restrict__ off, const int* __restrict__ padc,
                         float* __restrict__ sum){
  const int colTile = blockIdx.x * 64;
  int l = -1;
  for (int t = 0; t < NL; ++t)
    if (colTile >= off[t] && colTile < off[t] + padc[t]){ l = t; break; }
  if (l < 0) return;                      // beyond total padded width (uniform per block)
  const int tx = threadIdx.x;
  const int ch = blockIdx.y*4 + (tx >> 6);
  const int col = colTile + (tx & 63);
  __shared__ float red[256];
  red[tx] = bf2f(X[(size_t)ch*MP + col]);
  __syncthreads();
  for (int s = 32; s > 0; s >>= 1){
    if ((tx & 63) < s) red[tx] += red[tx + s];
    __syncthreads();
  }
  if ((tx & 63) == 0) atomicAdd(&sum[l*NCH + ch], red[tx]);
}

__global__ void k_mean(const float* __restrict__ sum, const int* __restrict__ cnt,
                       float* __restrict__ mu){
  int l = blockIdx.x, ch = threadIdx.x;
  mu[l*NCH + ch] = sum[l*NCH + ch] / fmaxf((float)cnt[l], 1.f);
}

// ---------------- per-label Gram via bf16 WMMA (register-staged pipeline) ----------------
// grid: (16 ij-blocks of 64x64, NL labels, KSPLIT), block = 256 (8 waves, 2 tiles/wave)
__global__ void __launch_bounds__(256)
k_gram(const unsigned short* __restrict__ X, float* __restrict__ gram,
       const int* __restrict__ off, const int* __restrict__ padc){
  const int iBlk = blockIdx.x >> 2, jBlk = blockIdx.x & 3;
  const int l = blockIdx.y, ks = blockIdx.z;
  const int base = off[l];
  const int chunks = padc[l] >> 5;                     // K-chunks of 32 pixels
  const int c0 = (chunks * ks) / KSPLIT, c1 = (chunks * (ks+1)) / KSPLIT;
  if (c0 >= c1) return;
  __shared__ __align__(16) unsigned short Asm[64*ASTR];
  __shared__ __align__(16) unsigned short Bsm[64*ASTR];
  const int tx = threadIdx.x;
  const int wave = tx >> 5, lane = tx & 31;
  const int t0 = wave*2, t1 = wave*2 + 1;              // 16 (16x16) tiles over 8 waves
  const int ti0 = t0 >> 2, tj0 = t0 & 3, ti1 = t1 >> 2, tj1 = t1 & 3;
  const int mrow = lane & 15, kb = (lane >> 4) * 8;    // fragment layout (ISA 7.12.2)
  const int lr = tx >> 2;                              // cooperative loader: row 0..63
  const int lc = (tx & 3) * 8;                         // 8 bf16 per thread
  const size_t rowA = (size_t)(iBlk*64 + lr)*MP;
  const size_t rowB = (size_t)(jBlk*64 + lr)*MP;
  v8f acc0 = {}; v8f acc1 = {};
  // pipeline: stage next chunk in VGPRs while WMMAs consume current LDS tile
  uint4 ra = *(const uint4*)&X[rowA + base + c0*32 + lc];
  uint4 rb = *(const uint4*)&X[rowB + base + c0*32 + lc];
  for (int c = c0; c < c1; ++c){
    __syncthreads();                                   // prior fragment reads done
    *(uint4*)&Asm[lr*ASTR + lc] = ra;
    *(uint4*)&Bsm[lr*ASTR + lc] = rb;
    __syncthreads();                                   // tiles visible
    if (c + 1 < c1){
      const int k0n = base + (c + 1)*32;
      ra = *(const uint4*)&X[rowA + k0n + lc];         // overlaps with WMMA below
      rb = *(const uint4*)&X[rowB + k0n + lc];
      __builtin_prefetch(&X[rowA + k0n + 32 + lc], 0, 1);
      __builtin_prefetch(&X[rowB + k0n + 32 + lc], 0, 1);
    }
    union FU { v16bf v; uint4 q[2]; };
    FU a0, b0, a1, b1;
    const unsigned short* pa0 = &Asm[(ti0*16 + mrow)*ASTR + kb];
    a0.q[0] = *(const uint4*)pa0; a0.q[1] = *(const uint4*)(pa0 + 16);
    const unsigned short* pb0 = &Bsm[(tj0*16 + mrow)*ASTR + kb];
    b0.q[0] = *(const uint4*)pb0; b0.q[1] = *(const uint4*)(pb0 + 16);
    const unsigned short* pa1 = &Asm[(ti1*16 + mrow)*ASTR + kb];
    a1.q[0] = *(const uint4*)pa1; a1.q[1] = *(const uint4*)(pa1 + 16);
    const unsigned short* pb1 = &Bsm[(tj1*16 + mrow)*ASTR + kb];
    b1.q[0] = *(const uint4*)pb1; b1.q[1] = *(const uint4*)(pb1 + 16);
    acc0 = __builtin_amdgcn_wmma_f32_16x16x32_bf16(false, a0.v, false, b0.v,
                                                   (short)0, acc0, false, false);
    acc1 = __builtin_amdgcn_wmma_f32_16x16x32_bf16(false, a1.v, false, b1.v,
                                                   (short)0, acc1, false, false);
  }
  float* G = gram + (size_t)l * NCH * NCH;
  const int n = lane & 15, mhi = (lane >> 4) * 8;
  #pragma unroll
  for (int r = 0; r < 8; ++r){
    atomicAdd(&G[(size_t)(iBlk*64 + ti0*16 + r + mhi)*NCH + (jBlk*64 + tj0*16 + n)], acc0[r]);
    atomicAdd(&G[(size_t)(iBlk*64 + ti1*16 + r + mhi)*NCH + (jBlk*64 + tj1*16 + n)], acc1[r]);
  }
}

// ---------------- Gram -> covariance in place ----------------
__global__ void k_cov(float* __restrict__ A, const float* __restrict__ mu,
                      const int* __restrict__ cnt){
  const int l = blockIdx.y, i = blockIdx.x, j = threadIdx.x;
  const float n = (float)cnt[l];
  const float den = fmaxf(fmaxf(n, 1.f) - 1.f, 1.f);
  const size_t o = (size_t)l*NCH*NCH + (size_t)i*NCH + j;
  A[o] = (A[o] - n * mu[l*NCH + i] * mu[l*NCH + j]) / den;
}

// ---------------- Cholesky (in place, lower) + optional inv(L) ----------------
__global__ void k_chol(float* __restrict__ Am, float* __restrict__ inv,
                       const int* __restrict__ valid){
  const int l = blockIdx.x;
  if (!valid[l]) return;
  float* A = Am + (size_t)l*NCH*NCH;
  __shared__ float col[NCH];
  __shared__ float sinvd;
  const int tx = threadIdx.x;
  for (int k = 0; k < NCH; ++k){
    if (tx == 0){
      float d = sqrtf(fmaxf(A[k*NCH + k], 1e-20f));
      A[k*NCH + k] = d; col[k] = d; sinvd = 1.f / d;
    }
    __syncthreads();
    if (tx > k){
      float v = A[tx*NCH + k] * sinvd;
      A[tx*NCH + k] = v; col[tx] = v;
    }
    __syncthreads();
    if (tx > k){
      const float ci = col[tx];
      for (int j = k + 1; j <= tx; ++j) A[tx*NCH + j] -= ci * col[j];
    }
    __syncthreads();
  }
  if (inv){                                   // forward substitution, column per thread
    float* V = inv + (size_t)l*NCH*NCH;
    const int j = tx;
    for (int i = 0; i < NCH; ++i){
      if (i < j){ V[i*NCH + j] = 0.f; continue; }
      float s = (i == j) ? 1.f : 0.f;
      for (int m = j; m < i; ++m) s -= A[i*NCH + m] * V[m*NCH + j];
      V[i*NCH + j] = s / A[i*NCH + i];
    }
  }
}

// ---------------- T = Ls * inv(Lc)  (both lower-tri => T lower-tri) ----------------
__global__ void k_combine(const float* __restrict__ Ls, const float* __restrict__ iLc,
                          float* __restrict__ T, unsigned short* __restrict__ Tbf,
                          const int* __restrict__ valid){
  const int l = blockIdx.x; if (!valid[l]) return;
  const float* B = Ls  + (size_t)l*NCH*NCH;
  const float* C = iLc + (size_t)l*NCH*NCH;
  float* O = T + (size_t)l*NCH*NCH;
  unsigned short* Ob = Tbf + (size_t)l*NCH*NCH;
  const int j = threadIdx.x;
  for (int i = 0; i < NCH; ++i){
    float s = 0.f;
    if (i >= j)
      for (int m = j; m <= i; ++m) s += B[i*NCH + m] * C[m*NCH + j];
    O[i*NCH + j] = s; Ob[i*NCH + j] = f2bf(s);
  }
}

__global__ void k_bias(const float* __restrict__ T, const float* __restrict__ mu_c,
                       const float* __restrict__ mu_s, float* __restrict__ bias,
                       const int* __restrict__ valid){
  const int l = blockIdx.x; if (!valid[l]) return;
  const int ch = threadIdx.x;
  const float* Tm = T + (size_t)l*NCH*NCH + (size_t)ch*NCH;
  const float* mc = mu_c + l*NCH;
  float s = 0.f;
  for (int j = 0; j < NCH; ++j) s += Tm[j] * mc[j];
  bias[l*NCH + ch] = mu_s[l*NCH + ch] - s;
}

// ---------------- apply: D = Xp^T tiles x T^T tiles = (T*X)^T, no LDS ----------------
// A-frag: 16 pixel rows of Xp (K = in-channels, contiguous).
// B-frag: 16 rows of T (row-major, K contiguous) -> B = T^T.
// grid: MP/64 column tiles; block = 256 (8 waves: 4 px-tiles x 2 n-halves, 8 n-tiles each)
__global__ void __launch_bounds__(256)
k_apply(const unsigned short* __restrict__ Xp, const unsigned short* __restrict__ Tbf,
        const float* __restrict__ bias,
        const int* __restrict__ off, const int* __restrict__ padc,
        const int* __restrict__ idx, const int* __restrict__ valid,
        float* __restrict__ out){
  const int colTile = blockIdx.x * 64;
  int l = -1;
  for (int t = 0; t < NL; ++t)
    if (colTile >= off[t] && colTile < off[t] + padc[t]){ l = t; break; }
  if (l < 0 || !valid[l]) return;          // invalid: content copy stays
  const int tx = threadIdx.x;
  const int wave = tx >> 5, lane = tx & 31;
  const int pt = wave >> 1;                // pixel tile 0..3 (16 pixels each)
  const int nh = wave & 1;                 // n-half: out-channel tiles nh*8 .. nh*8+7
  const int mrow = lane & 15, kb = (lane >> 4) * 8;
  const unsigned short* Arow = Xp + (size_t)(colTile + pt*16 + mrow)*NCH;
  const unsigned short* Tl   = Tbf + (size_t)l*NCH*NCH;
  v8f acc[8];
  v8f vzero = {};
  #pragma unroll
  for (int t = 0; t < 8; ++t) acc[t] = vzero;
  union FU { v16bf v; uint4 q[2]; };
  for (int kc = 0; kc < NCH/32; ++kc){
    FU a;
    const unsigned short* pa = Arow + kc*32 + kb;
    a.q[0] = *(const uint4*)pa; a.q[1] = *(const uint4*)(pa + 16);
    __builtin_prefetch(Arow + kc*32 + 32 + kb, 0, 1);
    #pragma unroll
    for (int t = 0; t < 8; ++t){
      const unsigned short* pb = Tl + (size_t)((nh*8 + t)*16 + mrow)*NCH + kc*32 + kb;
      FU b;
      b.q[0] = *(const uint4*)pb; b.q[1] = *(const uint4*)(pb + 16);
      acc[t] = __builtin_amdgcn_wmma_f32_16x16x32_bf16(false, a.v, false, b.v,
                                                       (short)0, acc[t], false, false);
    }
  }
  // epilogue: D rows = pixels, cols = out-channels; scatter by idx
  const int n = lane & 15, mhi = (lane >> 4) * 8;
  int prow[8];
  #pragma unroll
  for (int r = 0; r < 8; ++r) prow[r] = idx[colTile + pt*16 + r + mhi];
  #pragma unroll
  for (int t = 0; t < 8; ++t){
    const int outch = (nh*8 + t)*16 + n;
    const float bv = bias[l*NCH + outch];
    #pragma unroll
    for (int r = 0; r < 8; ++r){
      const int p = prow[r];
      if (p >= 0) out[(size_t)outch*HW + p] = acc[t][r] + bv;
    }
  }
}

// ================= host side =================
extern "C" void kernel_launch(void* const* d_in, const int* in_sizes, int n_in,
                              void* d_out, int out_size, void* d_ws, size_t ws_size,
                              hipStream_t stream){
  (void)in_sizes; (void)n_in; (void)out_size; (void)ws_size;
  const float* content = (const float*)d_in[0];
  const float* style   = (const float*)d_in[1];
  const int* seg_c = (const int*)d_in[2];
  const int* seg_s = (const int*)d_in[3];
  float* out = (float*)d_out;

  char* w = (char*)d_ws; size_t o = 0;
  auto alloc = [&](size_t bytes)->char*{
    char* p = w + o; o = (o + bytes + 255) & ~(size_t)255; return p;
  };
  int* ctrl = (int*)alloc(256 * sizeof(int));
  int *cnt_c = ctrl,      *cnt_s = ctrl + 16, *cur_c = ctrl + 32, *cur_s = ctrl + 48;
  int *off_c = ctrl + 64, *off_s = ctrl + 80, *padc_c = ctrl + 96, *padc_s = ctrl + 112;
  int *valid = ctrl + 128, *totals = ctrl + 144;
  float* sum_c = (float*)alloc(NL*NCH*sizeof(float));
  float* sum_s = (float*)alloc(NL*NCH*sizeof(float));
  float* mu_c  = (float*)alloc(NL*NCH*sizeof(float));
  float* mu_s  = (float*)alloc(NL*NCH*sizeof(float));
  float* gram_c = (float*)alloc((size_t)NL*NCH*NCH*sizeof(float));
  float* gram_s = (float*)alloc((size_t)NL*NCH*NCH*sizeof(float));
  float* iLc    = (float*)alloc((size_t)NL*NCH*NCH*sizeof(float));
  float* Tm     = (float*)alloc((size_t)NL*NCH*NCH*sizeof(float));
  float* bias   = (float*)alloc(NL*NCH*sizeof(float));
  unsigned short* Tbf = (unsigned short*)alloc((size_t)NL*NCH*NCH*sizeof(unsigned short));
  int* dst_c = (int*)alloc((size_t)HW*sizeof(int));
  int* dst_s = (int*)alloc((size_t)HW*sizeof(int));
  int* idx_c = (int*)alloc((size_t)MP*sizeof(int));
  unsigned short* Xc = (unsigned short*)alloc((size_t)NCH*MP*sizeof(unsigned short));
  unsigned short* Xs = (unsigned short*)alloc((size_t)NCH*MP*sizeof(unsigned short));
  unsigned short* Xp = (unsigned short*)alloc((size_t)NCH*MP*sizeof(unsigned short));

  hipMemsetAsync(ctrl, 0, 256*sizeof(int), stream);
  hipMemsetAsync(sum_c, 0, NL*NCH*sizeof(float), stream);
  hipMemsetAsync(sum_s, 0, NL*NCH*sizeof(float), stream);
  hipMemsetAsync(gram_c, 0, (size_t)NL*NCH*NCH*sizeof(float), stream);
  hipMemsetAsync(gram_s, 0, (size_t)NL*NCH*NCH*sizeof(float), stream);

  k_hist<<<1024, 256, 0, stream>>>(seg_c, cnt_c);
  k_hist<<<1024, 256, 0, stream>>>(seg_s, cnt_s);
  k_offsets<<<1, 32, 0, stream>>>(cnt_c, cnt_s, off_c, off_s, padc_c, padc_s, valid, totals);
  k_scatter<<<HW/256, 256, 0, stream>>>(seg_c, cur_c, off_c, dst_c, idx_c);
  k_scatter<<<HW/256, 256, 0, stream>>>(seg_s, cur_s, off_s, dst_s, (int*)nullptr);
  k_zeropad<<<NL, 256, 0, stream>>>(Xc, Xp, Xs, idx_c,
                                    cnt_c, off_c, padc_c, cnt_s, off_s, padc_s);
  k_gather<<<dim3(HW/256, NCH), 256, 0, stream>>>(content, dst_c, Xc, Xp);
  k_gather<<<dim3(HW/256, NCH), 256, 0, stream>>>(style,   dst_s, Xs, (unsigned short*)nullptr);
  k_colsum<<<dim3(MP/64, NCH/4), 256, 0, stream>>>(Xc, off_c, padc_c, sum_c);
  k_colsum<<<dim3(MP/64, NCH/4), 256, 0, stream>>>(Xs, off_s, padc_s, sum_s);
  k_mean<<<NL, NCH, 0, stream>>>(sum_c, cnt_c, mu_c);
  k_mean<<<NL, NCH, 0, stream>>>(sum_s, cnt_s, mu_s);
  k_gram<<<dim3(16, NL, KSPLIT), 256, 0, stream>>>(Xc, gram_c, off_c, padc_c);
  k_gram<<<dim3(16, NL, KSPLIT), 256, 0, stream>>>(Xs, gram_s, off_s, padc_s);
  k_cov<<<dim3(NCH, NL), NCH, 0, stream>>>(gram_c, mu_c, cnt_c);
  k_cov<<<dim3(NCH, NL), NCH, 0, stream>>>(gram_s, mu_s, cnt_s);
  k_chol<<<NL, NCH, 0, stream>>>(gram_c, iLc, valid);             // content: Lc + inv(Lc)
  k_chol<<<NL, NCH, 0, stream>>>(gram_s, (float*)nullptr, valid); // style: Ls
  k_combine<<<NL, NCH, 0, stream>>>(gram_s, iLc, Tm, Tbf, valid);
  k_bias<<<NL, NCH, 0, stream>>>(Tm, mu_c, mu_s, bias, valid);
  hipMemcpyAsync(out, content, (size_t)NCH*HW*sizeof(float),
                 hipMemcpyDeviceToDevice, stream);
  k_apply<<<MP/64, 256, 0, stream>>>(Xp, Tbf, bias, off_c, padc_c, idx_c, valid, out);
}
